// BahdanauAttention_2284922602060
// MI455X (gfx1250) — matrix operand order repaired and checked by reading, not compiled, not run on hardware
//
#include <hip/hip_runtime.h>
#include <hip/hip_bf16.h>
#include <math.h>

#define H_DIM 512
#define L_DEC 2
#define B_SZ 64
#define T_SEQ 2048
#define M_BLOCK 64        // rows of t per block (and per wave)
#define KSTEP 32
#define A_STRIDE 520      // 512 + 8 bf16 pad: row*1040B -> conflict-free ds_load_b128

typedef __attribute__((ext_vector_type(16))) __bf16 v16bf;
typedef __attribute__((ext_vector_type(8)))  float  v8f;

static __device__ __forceinline__ unsigned short f2bf_bits(float f) {
  unsigned u = __builtin_bit_cast(unsigned, f);
  u += 0x7FFFu + ((u >> 16) & 1u);        // round-to-nearest-even
  return (unsigned short)(u >> 16);
}

// ---------------------------------------------------------------- We -> bf16
__global__ __launch_bounds__(256) void convert_we_kernel(
    const float* __restrict__ We, unsigned short* __restrict__ WeB) {
  int i = blockIdx.x * 256 + threadIdx.x;
  if (i < H_DIM * H_DIM) WeB[i] = f2bf_bits(We[i]);
}

// -------------------------------------------- hidden_attn = mean_L(h)@Wh.T+bh
__global__ __launch_bounds__(256) void hidden_proj_kernel(
    const float* __restrict__ hidden,   // [L,B,H]
    const float* __restrict__ Wh,       // [H,H]
    const float* __restrict__ bh,       // [H]
    float* __restrict__ hatt) {         // [B,H]
  __shared__ __align__(16) float hm[H_DIM];
  const int b = blockIdx.x;
  for (int i = threadIdx.x; i < H_DIM; i += 256) {
    float s = 0.f;
    for (int l = 0; l < L_DEC; ++l)
      s += hidden[((size_t)l * B_SZ + b) * H_DIM + i];
    hm[i] = s * (1.0f / L_DEC);
  }
  __syncthreads();
  const float4* hv = reinterpret_cast<const float4*>(hm);
  for (int o = threadIdx.x; o < H_DIM; o += 256) {
    const float4* wv = reinterpret_cast<const float4*>(Wh + (size_t)o * H_DIM);
    float acc = 0.f;
#pragma unroll 4
    for (int i = 0; i < H_DIM / 4; ++i) {
      float4 w = wv[i], h4 = hv[i];
      acc = fmaf(w.x, h4.x, acc);
      acc = fmaf(w.y, h4.y, acc);
      acc = fmaf(w.z, h4.z, acc);
      acc = fmaf(w.w, h4.w, acc);
    }
    hatt[(size_t)b * H_DIM + o] = acc + bh[o];
  }
}

// ---- fused: scores[b,t] = sum_o Wo[o]*tanh(enc[b,t,:]@We[o,:] + be[o] + hatt[b,o])
// Block: 64 rows (t) x 512 channels (o). 8 waves; each wave = 64x64 output tile
// (4 M-tiles x 4 N-frags = 16 v8f accumulators), B frags reused 4x per K-step.
__global__ __launch_bounds__(256) void attn_scores_kernel(
    const float* __restrict__ enc,            // [T,B,H]
    const unsigned short* __restrict__ WeB,   // [H,H] bf16 bits, row-major (o, h)
    const float* __restrict__ be,
    const float* __restrict__ hatt,           // [B,H]
    const float* __restrict__ Wo,
    const float* __restrict__ bo,
    float* __restrict__ scores) {             // [B,T]
  extern __shared__ char smem[];
  unsigned short* Ash = reinterpret_cast<unsigned short*>(smem);              // 64*520*2 B
  float* sred = reinterpret_cast<float*>(smem + (size_t)M_BLOCK * A_STRIDE * 2); // 64 floats

  const int b   = blockIdx.y;
  const int t0  = blockIdx.x * M_BLOCK;
  const int tid = threadIdx.x;

  // Stage 64 x 512 enc tile into LDS as bf16 (coalesced fp32 reads).
  for (int idx = tid; idx < M_BLOCK * H_DIM; idx += 256) {
    int row = idx >> 9;
    int col = idx & (H_DIM - 1);
    float v = enc[((size_t)(t0 + row) * B_SZ + b) * H_DIM + col];
    Ash[row * A_STRIDE + col] = f2bf_bits(v);
  }
  if (tid < M_BLOCK) sred[tid] = 0.0f;
  __syncthreads();

  const int wave  = tid >> 5;       // 0..7, owns 64 output channels
  const int lane  = tid & 31;
  const int lhalf = lane >> 4;      // 0 | 1
  const int l16   = lane & 15;
  const int nbase = wave * 64;

  float addc[4], woc[4];
#pragma unroll
  for (int f = 0; f < 4; ++f) {
    int o = nbase + f * 16 + l16;
    addc[f] = be[o] + hatt[(size_t)b * H_DIM + o];
    woc[f]  = Wo[o];
  }

  v8f acc[4][4] = {};   // [m-tile][n-frag]

  // A fragment base (ISA 16-bit A 16x32 layout): lane half selects K sub-range.
  const unsigned short* arow = Ash + l16 * A_STRIDE + lhalf * 8;
  // B fragment bases: lane = column (o), K = lhalf*8 .. +8 and 16+lhalf*8 .. +8 (matches A).
  const unsigned short* bp[4];
#pragma unroll
  for (int f = 0; f < 4; ++f)
    bp[f] = WeB + (size_t)(nbase + f * 16 + l16) * H_DIM + lhalf * 8;

  union Frag { v16bf v; float4 q[2]; };

  for (int kk = 0; kk < H_DIM; kk += KSTEP) {
    Frag bf[4];
#pragma unroll
    for (int f = 0; f < 4; ++f) {
      bf[f].q[0] = *reinterpret_cast<const float4*>(bp[f] + kk);
      bf[f].q[1] = *reinterpret_cast<const float4*>(bp[f] + kk + 16);
    }
#pragma unroll
    for (int m = 0; m < 4; ++m) {
      Frag a;
      const unsigned short* ap = arow + m * 16 * A_STRIDE + kk;
      a.q[0] = *reinterpret_cast<const float4*>(ap);        // K = kk + lhalf*8 .. +8
      a.q[1] = *reinterpret_cast<const float4*>(ap + 16);   // K = kk+16+lhalf*8 .. +8
      acc[m][0] = __builtin_amdgcn_wmma_f32_16x16x32_bf16(false, a.v, false, bf[0].v, (short)0, acc[m][0], false, false);
      acc[m][1] = __builtin_amdgcn_wmma_f32_16x16x32_bf16(false, a.v, false, bf[1].v, (short)0, acc[m][1], false, false);
      acc[m][2] = __builtin_amdgcn_wmma_f32_16x16x32_bf16(false, a.v, false, bf[2].v, (short)0, acc[m][2], false, false);
      acc[m][3] = __builtin_amdgcn_wmma_f32_16x16x32_bf16(false, a.v, false, bf[3].v, (short)0, acc[m][3], false, false);
    }
  }

  // Epilogue: tanh + Wo-weighted reduction over the o (column) dimension.
  // C layout: VGPR r -> row (m*16 + r + lhalf*8), column = l16.
#pragma unroll
  for (int m = 0; m < 4; ++m) {
#pragma unroll
    for (int r = 0; r < 8; ++r) {
      float v = tanhf(acc[m][0][r] + addc[0]) * woc[0]
              + tanhf(acc[m][1][r] + addc[1]) * woc[1]
              + tanhf(acc[m][2][r] + addc[2]) * woc[2]
              + tanhf(acc[m][3][r] + addc[3]) * woc[3];
      v += __shfl_xor(v, 1, 32);
      v += __shfl_xor(v, 2, 32);
      v += __shfl_xor(v, 4, 32);
      v += __shfl_xor(v, 8, 32);   // masks <=8 stay within each 16-lane half
      if (l16 == 0)
        atomicAdd(&sred[m * 16 + r + lhalf * 8], v);   // ds_add_f32 across 8 waves
    }
  }
  __syncthreads();
  if (tid < M_BLOCK)
    scores[(size_t)b * T_SEQ + t0 + tid] = sred[tid] + bo[0];
}

// --------------------------------------------------- per-batch masked softmax
__global__ __launch_bounds__(256) void softmax_kernel(
    const float* __restrict__ scores, const int* __restrict__ enc_len,
    float* __restrict__ out) {
  __shared__ float red[8];
  const int b   = blockIdx.x;
  const int tid = threadIdx.x;
  int len = enc_len[b];
  len = len < 0 ? 0 : (len > T_SEQ ? T_SEQ : len);

  float v[T_SEQ / 256];
  float m = -__builtin_inff();
#pragma unroll
  for (int i = 0; i < T_SEQ / 256; ++i) {
    int t = tid + i * 256;
    v[i] = (t < len) ? scores[(size_t)b * T_SEQ + t] : -__builtin_inff();
    m = fmaxf(m, v[i]);
  }
  for (int off = 16; off >= 1; off >>= 1) m = fmaxf(m, __shfl_xor(m, off, 32));
  if ((tid & 31) == 0) red[tid >> 5] = m;
  __syncthreads();
  m = red[0];
#pragma unroll
  for (int w = 1; w < 8; ++w) m = fmaxf(m, red[w]);
  __syncthreads();

  float ssum = 0.f;
#pragma unroll
  for (int i = 0; i < T_SEQ / 256; ++i) {
    int t = tid + i * 256;
    float e = (t < len) ? __expf(v[i] - m) : 0.f;
    v[i] = e;
    ssum += e;
  }
  for (int off = 16; off >= 1; off >>= 1) ssum += __shfl_xor(ssum, off, 32);
  if ((tid & 31) == 0) red[tid >> 5] = ssum;
  __syncthreads();
  ssum = 0.f;
#pragma unroll
  for (int w = 0; w < 8; ++w) ssum += red[w];
  float inv = 1.0f / ssum;
#pragma unroll
  for (int i = 0; i < T_SEQ / 256; ++i) {
    int t = tid + i * 256;
    out[(size_t)b * T_SEQ + t] = v[i] * inv;   // masked entries were set to 0
  }
}

extern "C" void kernel_launch(void* const* d_in, const int* in_sizes, int n_in,
                              void* d_out, int out_size, void* d_ws, size_t ws_size,
                              hipStream_t stream) {
  const float* hidden  = (const float*)d_in[0];   // [L,B,H]
  const float* enc     = (const float*)d_in[1];   // [T,B,H]
  const int*   enc_len = (const int*)  d_in[2];   // [B]
  const float* Wh      = (const float*)d_in[3];   // [H,H]
  const float* bh      = (const float*)d_in[4];   // [H]
  const float* We      = (const float*)d_in[5];   // [H,H]
  const float* be      = (const float*)d_in[6];   // [H]
  const float* Wo      = (const float*)d_in[7];   // [H]
  const float* bo      = (const float*)d_in[8];   // [1]
  float* out = (float*)d_out;                     // [B,T] (== [B,T,1])

  char* ws = (char*)d_ws;
  unsigned short* WeB  = (unsigned short*)ws;                       // 512 KB
  float*          hatt = (float*)(ws + 524288);                     // 128 KB
  float*          scr  = (float*)(ws + 524288 + 131072);            // 512 KB

  convert_we_kernel<<<(H_DIM * H_DIM) / 256, 256, 0, stream>>>(We, WeB);
  hidden_proj_kernel<<<B_SZ, 256, 0, stream>>>(hidden, Wh, bh, hatt);

  const size_t smem_bytes = (size_t)M_BLOCK * A_STRIDE * 2 + M_BLOCK * sizeof(float);
  dim3 grid(T_SEQ / M_BLOCK, B_SZ);  // 32 x 64 = 2048 blocks
  attn_scores_kernel<<<grid, 256, smem_bytes, stream>>>(enc, WeB, be, hatt, Wo, bo, scr);

  softmax_kernel<<<B_SZ, 256, 0, stream>>>(scr, enc_len, out);
}
